// NodeGCN_19808389169615
// MI455X (gfx1250) — matrix operand order repaired and checked by reading, not compiled, not run on hardware
//
#include <hip/hip_runtime.h>

typedef __attribute__((ext_vector_type(16))) _Float16 v16h;
typedef __attribute__((ext_vector_type(8)))  float    v8f;

#define GWAVES 4   // waves per block; block = 128 threads = 4 x wave32

// A-fragment K index for 16-bit WMMA: lanes 0-15 hold K {0..15},
// lanes 16-31 hold K {16..31} (contiguous, matching the B fragment fills).
__device__ __forceinline__ int a_frag_k(int e, int hi8) {
  return e + 2 * hi8;
}

// ---------------------------------------------------------------------------
// Generic fused GEMM: Y[m, 0:NC] = act(X[m, 0:K] @ W + B)
//   W is [K, NC] row-major in global, stored column-major f16 in LDS so that
//   B-fragment gathers are contiguous 16-half runs.
//   K % 32 == 0, NC % 16 == 0. One wave computes a 16-row x NC tile.
// ---------------------------------------------------------------------------
template<int K, int NC>
__global__ void __launch_bounds__(32 * GWAVES)
gemm_wmma(const float* __restrict__ X, long ldx,
          const float* __restrict__ W, const float* __restrict__ B,
          float* __restrict__ Y, long ldy, int n, int act)
{
  constexpr int NT = NC / 16;
  constexpr int KC = K / 32;
  __shared__ _Float16 sW[NC * K];            // column-major: [c*K + k]
  __shared__ float    sB[NC];
  __shared__ _Float16 sA[GWAVES][16 * 32];   // per-wave A tile, row-major 16x32

  const int tid  = threadIdx.x;
  const int wave = tid >> 5;
  const int lane = tid & 31;
  const int l15  = lane & 15;
  const int hi8  = (lane >= 16) ? 8 : 0;
  const int hi16 = (lane >= 16) ? 16 : 0;

  for (int i = tid; i < K * NC; i += 32 * GWAVES) {
    int c = i / K, k = i % K;
    sW[i] = (_Float16)W[(long)k * NC + c];
  }
  for (int i = tid; i < NC; i += 32 * GWAVES) sB[i] = B[i];
  __syncthreads();

  const int row0 = (blockIdx.x * GWAVES + wave) * 16;
  _Float16* myA = &sA[wave][0];

  v8f acc[NT];
  #pragma unroll
  for (int nt = 0; nt < NT; ++nt) {
    float bv = sB[nt * 16 + l15];
    #pragma unroll
    for (int r = 0; r < 8; ++r) acc[nt][r] = bv;
  }

  for (int kc = 0; kc < KC; ++kc) {
    // cooperative, coalesced A-tile load (rows row0..+15, cols kc*32..+31)
    #pragma unroll
    for (int j = 0; j < 16; ++j) {
      int r = row0 + j;
      float v = (r < n) ? X[(long)r * ldx + kc * 32 + lane] : 0.0f;
      myA[j * 32 + lane] = (_Float16)v;
    }
    if (kc + 1 < KC) {
      int pr = row0 + l15; if (pr >= n) pr = n - 1;
      __builtin_prefetch(&X[(long)pr * ldx + (kc + 1) * 32 + lane], 0, 1);
    }
    v16h af;
    #pragma unroll
    for (int e = 0; e < 16; ++e) af[e] = myA[l15 * 32 + a_frag_k(e, hi8)];
    #pragma unroll
    for (int nt = 0; nt < NT; ++nt) {
      v16h bf;
      #pragma unroll
      for (int e = 0; e < 16; ++e)
        bf[e] = sW[(nt * 16 + l15) * K + kc * 32 + e + hi16];
      acc[nt] = __builtin_amdgcn_wmma_f32_16x16x32_f16(
          false, af, false, bf, (short)0, acc[nt], false, false);
    }
  }

  #pragma unroll
  for (int nt = 0; nt < NT; ++nt) {
    #pragma unroll
    for (int r = 0; r < 8; ++r) {
      int m = row0 + r + hi8;                 // D layout: M = r + 8*hi, N = lane&15
      if (m < n) {
        float v = acc[nt][r];
        if (act) v = fmaxf(v, 0.0f);
        Y[(long)m * ldy + nt * 16 + l15] = v;
      }
    }
  }
}

// ---------------------------------------------------------------------------
// Fused edge MLP + message + scatter:
//   ee = relu(eattr @ W1 + B1) @ W2 + B2          (WMMA, per 16-edge tile)
//   msg = norm * relu(hprev[row] + ee)
//   atomicAdd(agg[col], msg)
// eattr [E,16]; W1 [16,96]; W2 [96,96]; hprev read with (ldh, hoff).
// ---------------------------------------------------------------------------
__global__ void __launch_bounds__(32 * GWAVES)
edge_mlp_scatter(const float* __restrict__ eattr,
                 const int*   __restrict__ eidx,     // [2,E]: row then col
                 const float* __restrict__ W1, const float* __restrict__ B1,
                 const float* __restrict__ W2, const float* __restrict__ B2,
                 const float* __restrict__ hprev, long ldh, int hoff,
                 const float* __restrict__ dinv, int use_norm,
                 float* __restrict__ agg, int E)
{
  __shared__ _Float16 sW1[96 * 32];           // column-major [c*32+k], K padded 16->32
  __shared__ _Float16 sW2[96 * 96];           // column-major [c*96+k]
  __shared__ float    sB1[96], sB2[96];
  __shared__ _Float16 sA[GWAVES][16 * 32];    // edge_attr tile, padded
  __shared__ _Float16 sT[GWAVES][16 * 96];    // relu hidden, f16

  const int tid  = threadIdx.x;
  const int wave = tid >> 5;
  const int lane = tid & 31;
  const int l15  = lane & 15;
  const int hi8  = (lane >= 16) ? 8 : 0;
  const int hi16 = (lane >= 16) ? 16 : 0;

  for (int i = tid; i < 96 * 32; i += 32 * GWAVES) {
    int c = i / 32, k = i % 32;
    sW1[i] = (k < 16) ? (_Float16)W1[k * 96 + c] : (_Float16)0.0f;
  }
  for (int i = tid; i < 96 * 96; i += 32 * GWAVES) {
    int c = i / 96, k = i % 96;
    sW2[i] = (_Float16)W2[k * 96 + c];
  }
  for (int i = tid; i < 96; i += 32 * GWAVES) { sB1[i] = B1[i]; sB2[i] = B2[i]; }
  __syncthreads();

  const int ebase = (blockIdx.x * GWAVES + wave) * 16;
  _Float16* myA = &sA[wave][0];
  _Float16* myT = &sT[wave][0];

  // ---- stage 0: load 16x16 edge_attr tile (zero-padded to K=32) ----
  #pragma unroll
  for (int j = 0; j < 16; ++j) {
    int e = ebase + j;
    float v = (lane < 16 && e < E) ? eattr[(long)e * 16 + lane] : 0.0f;
    myA[j * 32 + lane] = (_Float16)v;
  }

  // ---- stage 1: T = relu(A @ W1 + B1), 16x96, store f16 to LDS ----
  {
    v16h af;
    #pragma unroll
    for (int e = 0; e < 16; ++e) af[e] = myA[l15 * 32 + a_frag_k(e, hi8)];
    #pragma unroll
    for (int nt = 0; nt < 6; ++nt) {
      v8f c;
      float bv = sB1[nt * 16 + l15];
      #pragma unroll
      for (int r = 0; r < 8; ++r) c[r] = bv;
      v16h bf;
      #pragma unroll
      for (int e = 0; e < 16; ++e)
        bf[e] = sW1[(nt * 16 + l15) * 32 + e + hi16];
      c = __builtin_amdgcn_wmma_f32_16x16x32_f16(false, af, false, bf,
                                                 (short)0, c, false, false);
      #pragma unroll
      for (int r = 0; r < 8; ++r) {
        int m = r + hi8;
        myT[m * 96 + nt * 16 + l15] = (_Float16)fmaxf(c[r], 0.0f);
      }
    }
  }
  __syncthreads();

  // ---- stage 2: ee = T @ W2 + B2 (K=96), accumulators stay in VGPRs ----
  v8f acc[6];
  #pragma unroll
  for (int nt = 0; nt < 6; ++nt) {
    float bv = sB2[nt * 16 + l15];
    #pragma unroll
    for (int r = 0; r < 8; ++r) acc[nt][r] = bv;
  }
  #pragma unroll
  for (int kc = 0; kc < 3; ++kc) {
    v16h af;
    #pragma unroll
    for (int e = 0; e < 16; ++e)
      af[e] = myT[l15 * 96 + kc * 32 + a_frag_k(e, hi8)];
    #pragma unroll
    for (int nt = 0; nt < 6; ++nt) {
      v16h bf;
      #pragma unroll
      for (int e = 0; e < 16; ++e)
        bf[e] = sW2[(nt * 16 + l15) * 96 + kc * 32 + e + hi16];
      acc[nt] = __builtin_amdgcn_wmma_f32_16x16x32_f16(
          false, af, false, bf, (short)0, acc[nt], false, false);
    }
  }

  // ---- stage 3: consume D fragments in place: msg + atomic scatter ----
  #pragma unroll
  for (int nt = 0; nt < 6; ++nt) {
    int col = nt * 16 + l15;
    #pragma unroll
    for (int r = 0; r < 8; ++r) {
      int e = ebase + r + hi8;                // D layout: M = edge-in-tile
      if (e < E) {
        int src = eidx[e];
        int dst = eidx[E + e];
        float nrm = use_norm ? (dinv[src] * dinv[dst]) : 1.0f;
        float hv  = hprev[(long)src * ldh + hoff + col];
        float msg = fmaxf(hv + acc[nt][r], 0.0f) * nrm;
        atomicAdd(&agg[(long)dst * 96 + col], msg);
      }
    }
  }
}

// ----------------------------- elementwise ---------------------------------
__global__ void fill_f32(float* __restrict__ p, float v, long c) {
  long i = (long)blockIdx.x * 256 + threadIdx.x;
  if (i < c) p[i] = v;
}
__global__ void deg_accum(const int* __restrict__ eidx, float* __restrict__ deg, int E) {
  int i = blockIdx.x * 256 + threadIdx.x;
  if (i < E) atomicAdd(&deg[eidx[i]], 1.0f);
}
__global__ void rsqrt_k(const float* __restrict__ deg, float* __restrict__ dinv, int n) {
  int i = blockIdx.x * 256 + threadIdx.x;
  if (i < n) dinv[i] = rsqrtf(deg[i]);
}
// h = BN(relu(agg + relu(hlin + root)/deg)), write into feats[:, 0:96]
__global__ void gcn_combine(const float* __restrict__ hlin, const float* __restrict__ agg,
                            const float* __restrict__ root, const float* __restrict__ deg,
                            const float* __restrict__ g, const float* __restrict__ b,
                            const float* __restrict__ m, const float* __restrict__ v,
                            float* __restrict__ feats, int n) {
  long i = (long)blockIdx.x * 256 + threadIdx.x;
  if (i < (long)n * 96) {
    int node = (int)(i / 96), c = (int)(i % 96);
    float val = agg[i] + fmaxf(hlin[i] + root[c], 0.0f) / deg[node];
    val = fmaxf(val, 0.0f);
    val = (val - m[c]) * rsqrtf(v[c] + 1e-5f) * g[c] + b[c];
    feats[(long)node * 288 + c] = val;
  }
}
// z = (1+eps)*feats[:, hoff:hoff+96] + agg
__global__ void gin_z(const float* __restrict__ feats, int hoff,
                      const float* __restrict__ agg,
                      const float* __restrict__ eps, int li,
                      float* __restrict__ z, int n) {
  long i = (long)blockIdx.x * 256 + threadIdx.x;
  if (i < (long)n * 96) {
    int node = (int)(i / 96), c = (int)(i % 96);
    float s = 1.0f + eps[li];
    z[i] = s * feats[(long)node * 288 + hoff + c] + agg[i];
  }
}
// feats[:, hoff:hoff+96] = BN(relu(t))
__global__ void bn_relu(const float* __restrict__ t,
                        const float* __restrict__ g, const float* __restrict__ b,
                        const float* __restrict__ m, const float* __restrict__ v,
                        float* __restrict__ feats, int hoff, int n) {
  long i = (long)blockIdx.x * 256 + threadIdx.x;
  if (i < (long)n * 96) {
    int node = (int)(i / 96), c = (int)(i % 96);
    float val = fmaxf(t[i], 0.0f);
    val = (val - m[c]) * rsqrtf(v[c] + 1e-5f) * g[c] + b[c];
    feats[(long)node * 288 + hoff + c] = val;
  }
}

// ---------------------------------------------------------------------------
extern "C" void kernel_launch(void* const* d_in, const int* in_sizes, int n_in,
                              void* d_out, int out_size, void* d_ws, size_t ws_size,
                              hipStream_t stream) {
  const float* x       = (const float*)d_in[0];
  const int*   eidx    = (const int*)  d_in[1];
  const float* eattr   = (const float*)d_in[2];
  const float* lin_w   = (const float*)d_in[3];
  const float* lin_b   = (const float*)d_in[4];
  const float* root    = (const float*)d_in[5];
  const float* gcn_bw1 = (const float*)d_in[6];
  const float* gcn_bb1 = (const float*)d_in[7];
  const float* gcn_bw2 = (const float*)d_in[8];
  const float* gcn_bb2 = (const float*)d_in[9];
  const float* gin_bw1 = (const float*)d_in[10];
  const float* gin_bb1 = (const float*)d_in[11];
  const float* gin_bw2 = (const float*)d_in[12];
  const float* gin_bb2 = (const float*)d_in[13];
  const float* gin_mw1 = (const float*)d_in[14];
  const float* gin_mb1 = (const float*)d_in[15];
  const float* gin_mw2 = (const float*)d_in[16];
  const float* gin_mb2 = (const float*)d_in[17];
  const float* gin_eps = (const float*)d_in[18];
  const float* bn_g    = (const float*)d_in[19];
  const float* bn_b    = (const float*)d_in[20];
  const float* bn_m    = (const float*)d_in[21];
  const float* bn_v    = (const float*)d_in[22];
  const float* fc1_w   = (const float*)d_in[23];
  const float* fc1_b   = (const float*)d_in[24];
  const float* fc2_w   = (const float*)d_in[25];
  const float* fc2_b   = (const float*)d_in[26];

  const int H = 96;
  const int n = in_sizes[0] / 128;     // N_NODES
  const int E = in_sizes[2] / 16;      // N_EDGES

  float* deg   = (float*)d_ws;
  float* dinv  = deg  + n;
  float* tmp   = dinv + n;                       // [n, 96]
  float* agg   = tmp  + (size_t)n * H;           // [n, 96]
  float* feats = agg  + (size_t)n * H;           // [n, 288]
  float* out   = (float*)d_out;

  auto eb = [](long c) { return dim3((unsigned)((c + 255) / 256)); };
  dim3 b256(256), b128(32 * GWAVES);
  dim3 ggrid((unsigned)((n + 16 * GWAVES - 1) / (16 * GWAVES)));
  dim3 egrid((unsigned)((E + 16 * GWAVES - 1) / (16 * GWAVES)));
  long nh = (long)n * H;

  // degree / norm
  fill_f32<<<eb(n), b256, 0, stream>>>(deg, 1.0f, n);
  deg_accum<<<eb(E), b256, 0, stream>>>(eidx, deg, E);
  rsqrt_k<<<eb(n), b256, 0, stream>>>(deg, dinv, n);

  // ---- layer 0: GCN ----
  gemm_wmma<128, 96><<<ggrid, b128, 0, stream>>>(x, 128, lin_w, lin_b, tmp, 96, n, 0);
  fill_f32<<<eb(nh), b256, 0, stream>>>(agg, 0.0f, nh);
  edge_mlp_scatter<<<egrid, b128, 0, stream>>>(eattr, eidx, gcn_bw1, gcn_bb1,
                                               gcn_bw2, gcn_bb2,
                                               tmp, 96, 0, dinv, 1, agg, E);
  gcn_combine<<<eb(nh), b256, 0, stream>>>(tmp, agg, root, deg,
                                           bn_g, bn_b, bn_m, bn_v, feats, n);

  // ---- layers 1..2: GIN ----
  for (int i = 0; i < 2; ++i) {
    fill_f32<<<eb(nh), b256, 0, stream>>>(agg, 0.0f, nh);
    edge_mlp_scatter<<<egrid, b128, 0, stream>>>(eattr, eidx,
        gin_bw1 + (size_t)i * 16 * 96, gin_bb1 + (size_t)i * 96,
        gin_bw2 + (size_t)i * 96 * 96, gin_bb2 + (size_t)i * 96,
        feats, 288, i * 96, dinv, 0, agg, E);
    gin_z<<<eb(nh), b256, 0, stream>>>(feats, i * 96, agg, gin_eps, i, tmp, n);
    gemm_wmma<96, 96><<<ggrid, b128, 0, stream>>>(tmp, 96,
        gin_mw1 + (size_t)i * 96 * 96, gin_mb1 + (size_t)i * 96, agg, 96, n, 1);
    gemm_wmma<96, 96><<<ggrid, b128, 0, stream>>>(agg, 96,
        gin_mw2 + (size_t)i * 96 * 96, gin_mb2 + (size_t)i * 96, tmp, 96, n, 0);
    bn_relu<<<eb(nh), b256, 0, stream>>>(tmp,
        bn_g + (size_t)(i + 1) * 96, bn_b + (size_t)(i + 1) * 96,
        bn_m + (size_t)(i + 1) * 96, bn_v + (size_t)(i + 1) * 96,
        feats, (i + 1) * 96, n);
  }

  // ---- head ----
  gemm_wmma<288, 96><<<ggrid, b128, 0, stream>>>(feats, 288, fc1_w, fc1_b, tmp, 96, n, 1);
  gemm_wmma<96, 64><<<ggrid, b128, 0, stream>>>(tmp, 96, fc2_w, fc2_b, out, 64, n, 0);

  (void)n_in; (void)out_size; (void)ws_size;
  (void)gin_bb1; // (used above; silence any over-eager warnings)
}
